// RecurrentLevelUnit_21990232556152
// MI455X (gfx1250) — compile-verified, matrix operand order checked by reading, not run on hardware
//
#include <hip/hip_runtime.h>
#include <math.h>

// ---------------- constants (match reference) ----------------
constexpr int BB   = 2;    // batch
constexpr int CC   = 32;   // feature channels
constexpr int HH   = 64;
constexpr int WW   = 96;
constexpr int HWSZ = HH * WW;
constexpr int HIDC = 128;
constexpr int RR   = 3;
constexpr int DUU  = 7;
constexpr int NDISP = DUU * DUU;       // 49
constexpr int NITEMS = BB * NDISP;     // 98
constexpr int NCH  = 14;               // chunk size over items
constexpr int NCHUNKS = NITEMS / NCH;  // 7

typedef __bf16 bf16_t;
typedef bf16_t v16bf __attribute__((ext_vector_type(16)));
typedef float  v8f   __attribute__((ext_vector_type(8)));

union ABu { v16bf v; uint4 q[2]; };

// async global->LDS path (ASYNCcnt) if the toolchain exposes it
#if defined(__has_builtin)
#  if __has_builtin(__builtin_amdgcn_global_load_async_to_lds_b128) && \
      __has_builtin(__builtin_amdgcn_s_wait_asynccnt)
#    define USE_ASYNC_LDS 1
#  endif
#endif

#ifdef USE_ASYNC_LDS
typedef int v4i_t __attribute__((ext_vector_type(4)));
typedef __attribute__((address_space(1))) v4i_t gv4i_t;  // global src
typedef __attribute__((address_space(3))) v4i_t lv4i_t;  // LDS dst
#endif

__device__ __forceinline__ unsigned short f2bf(float f) {
    unsigned int u = __float_as_uint(f);
    unsigned int r = u + 0x7FFFu + ((u >> 16) & 1u);
    return (unsigned short)(r >> 16);
}

// ============================================================
// Implicit-GEMM conv via WMMA bf16.
//  - one 8-wave block owns a 16-pixel tile (one output row segment)
//  - per tap, the 16 x CIN bf16 A-tile is staged in LDS; with the async
//    path this is double-buffered: tap t+1 streams in via
//    GLOBAL_LOAD_ASYNC_TO_LDS_B128 (ASYNCcnt) under tap t's WMMAs
//  - each wave covers oc-tiles {w, w+8}, K-loop fully unrolled (CIN template)
//   act : [N][Hp][Wp][CIN]  bf16 (spatially pre-padded, CIN % 32 == 0)
//   wgt : [T][Opad][CIN]    bf16
//   out : bf16 NHWC (mode 0) or f32 NCHW planes (mode 1)
// ============================================================
template<int CIN>
__global__ __launch_bounds__(256) void conv_wmma_kernel(
    const unsigned short* __restrict__ act,
    const unsigned short* __restrict__ wgt,
    const float* __restrict__ bias,
    void* __restrict__ outp,
    int N, int Hp, int Wp,
    int Ho, int Wo, int stride, int iy0, int ix0,
    int Opad, int Oreal, int T,
    unsigned long long dys, unsigned long long dxs,
    int act_code, int out_mode,
    int oHp, int oWp, int oCt, int oy0, int ox0, int oys, int oxs, int ocoff)
{
    constexpr int CB  = CIN / 32;   // K-chunks of 32
    constexpr int SEG = CIN / 8;    // uint4 segments per pixel row
#ifdef USE_ASYNC_LDS
    __shared__ unsigned short lds_a[2][16 * CIN];
#else
    __shared__ unsigned short lds_a[1][16 * CIN];
#endif

    int nxt = Wo >> 4;
    int blk = blockIdx.x;                 // grid sized exactly N*Ho*nxt
    int xt = blk % nxt; int t1 = blk / nxt;
    int y  = t1 % Ho;   int n  = t1 / Ho;

    int noc   = Opad >> 4;
    int w     = threadIdx.x >> 5;
    int lane  = threadIdx.x & 31;
    int mrow  = lane & 15;
    int half8 = (lane & 16) >> 1;         // 0 or 8 (K-split halves, A/B layout)
    int xbase = xt << 4;

    int  ot0 = w, ot1 = w + 8;
    bool has0 = ot0 < noc, has1 = ot1 < noc;
    int  oc0 = (ot0 << 4) + mrow, oc1 = (ot1 << 4) + mrow;

    v8f acc0, acc1;
#pragma unroll
    for (int i = 0; i < 8; ++i) { acc0[i] = 0.0f; acc1[i] = 0.0f; }

    // cooperative stage of the 16 x CIN A-tile for tap t into buffer `buf`
    auto stage = [&](int t, int buf) {
        int dy = (int)((dys >> (4 * t)) & 0xFull); dy = (dy << 28) >> 28;
        int dx = (int)((dxs >> (4 * t)) & 0xFull); dx = (dx << 28) >> 28;
        int yin = y * stride + iy0 + dy;
        for (int idx = threadIdx.x; idx < 16 * SEG; idx += 256) {
            int row = idx / SEG; int seg = idx - row * SEG;
            int xin = (xbase + row) * stride + ix0 + dx;
            const unsigned short* src =
                act + ((size_t)(n * Hp + yin) * Wp + xin) * CIN + seg * 8;
            unsigned short* dst = &lds_a[buf][row * CIN + seg * 8];
#ifdef USE_ASYNC_LDS
            __builtin_amdgcn_global_load_async_to_lds_b128(
                (gv4i_t*)src, (lv4i_t*)dst, 0, 0);
#else
            *((uint4*)dst) = *(const uint4*)src;
#endif
        }
    };

#ifdef USE_ASYNC_LDS
    stage(0, 0);
#endif
    for (int t = 0; t < T; ++t) {
#ifdef USE_ASYNC_LDS
        int cur = t & 1;
        __builtin_amdgcn_s_wait_asynccnt(0);  // my LDS writes for buf[cur] done
        __syncthreads();                      // everyone's writes visible; buf[1-cur] free
        if (t + 1 < T) stage(t + 1, 1 - cur); // async stream next tap under the MACs
#else
        int cur = 0;
        __syncthreads();                      // previous iteration's LDS consumers done
        stage(t, 0);
        __syncthreads();                      // A-tile visible to all waves
#endif
        const unsigned short* arow  = &lds_a[cur][mrow * CIN] + half8;
        const unsigned short* brow0 = wgt + ((size_t)t * Opad + oc0) * CIN + half8;
        const unsigned short* brow1 = wgt + ((size_t)t * Opad + oc1) * CIN + half8;
        if (t + 1 < T) {  // prefetch next tap's weight rows (global_prefetch)
            __builtin_prefetch(wgt + ((size_t)(t + 1) * Opad + oc0) * CIN, 0, 1);
        }
#pragma unroll
        for (int c = 0; c < CB; ++c) {
            ABu a;
            a.q[0] = *(const uint4*)(arow + c * 32);
            a.q[1] = *(const uint4*)(arow + c * 32 + 16);
            if (has0) {
                ABu b;
                b.q[0] = *(const uint4*)(brow0 + c * 32);
                b.q[1] = *(const uint4*)(brow0 + c * 32 + 16);
                acc0 = __builtin_amdgcn_wmma_f32_16x16x32_bf16(
                    false, a.v, false, b.v, (short)0, acc0, false, false);
            }
            if (has1) {
                ABu b;
                b.q[0] = *(const uint4*)(brow1 + c * 32);
                b.q[1] = *(const uint4*)(brow1 + c * 32 + 16);
                acc1 = __builtin_amdgcn_wmma_f32_16x16x32_bf16(
                    false, a.v, false, b.v, (short)0, acc1, false, false);
            }
        }
    }

    unsigned short* outb = (unsigned short*)outp;
    float* outf = (float*)outp;
#pragma unroll
    for (int j = 0; j < 2; ++j) {
        if (j == 0 && !has0) continue;
        if (j == 1 && !has1) continue;
        int oc = j ? oc1 : oc0;
        bool wr = (oc < Oreal);
        if (!wr) continue;
        float bv = bias ? bias[oc] : 0.0f;
        v8f acc = j ? acc1 : acc0;
#pragma unroll
        for (int r = 0; r < 8; ++r) {
            float v = acc[r] + bv;
            if      (act_code == 1) v = v > 0.0f ? v : 0.0f;
            else if (act_code == 2) v = v > 0.0f ? v : 0.01f * v;
            else if (act_code == 3) v = 1.0f / (1.0f + __expf(-v));
            else if (act_code == 4) v = tanhf(v);
            int xo = xbase + r + half8;         // C/D layout: M = r + 8*(lane>=16)
            int yy = y * oys + oy0;
            int xx = xo * oxs + ox0;
            if (out_mode == 0) {
                outb[((size_t)(n * oHp + yy) * oWp + xx) * oCt + ocoff + oc] = f2bf(v);
            } else {
                outf[((size_t)n * oCt + (ocoff + oc)) * ((size_t)oHp * oWp)
                     + (size_t)yy * oWp + xx] = v;
            }
        }
    }
}

// ---------------- weight repack: OIHW f32 -> [t][o][ci] bf16 ----------------
__global__ void repack_w_kernel(const float* __restrict__ w, unsigned short* __restrict__ out,
                                int O, int I, int KH, int KW, int Opad, int Ipad, int T,
                                unsigned long long khs, unsigned long long kws, int total)
{
    int tid = blockIdx.x * blockDim.x + threadIdx.x;
    if (tid >= total) return;
    int ci = tid % Ipad; int t1 = tid / Ipad;
    int o  = t1 % Opad;  int t  = t1 / Opad;
    int kh = (int)((khs >> (4 * t)) & 0xFull);
    int kw = (int)((kws >> (4 * t)) & 0xFull);
    float v = 0.0f;
    if (o < O && ci < I) v = w[(((size_t)o * I + ci) * KH + kh) * KW + kw];
    out[tid] = f2bf(v);
}

// ---------------- bilinear backward warp into 3-padded buffer ----------------
__global__ void warp_pack_kernel(const float* __restrict__ fmap2, const float* __restrict__ flow,
                                 float* __restrict__ f2wp)
{
    int tid = blockIdx.x * blockDim.x + threadIdx.x;
    if (tid >= BB * HWSZ) return;
    int x = tid % WW; int y = (tid / WW) % HH; int b = tid / HWSZ;
    size_t p = (size_t)y * WW + x;
    float u = flow[((size_t)(b * 2 + 0)) * HWSZ + p];
    float v = flow[((size_t)(b * 2 + 1)) * HWSZ + p];
    float xs = (float)x + u, ys = (float)y + v;
    float x0f = floorf(xs), y0f = floorf(ys);
    float wx = xs - x0f, wy = ys - y0f;
    int x0 = (int)x0f, y0 = (int)y0f;
    float w00 = (1 - wy) * (1 - wx), w01 = (1 - wy) * wx;
    float w10 = wy * (1 - wx),       w11 = wy * wx;
    for (int c = 0; c < CC; ++c) {
        const float* fb = fmap2 + (size_t)(b * CC + c) * HWSZ;
        float acc = 0.0f;
        if (y0 >= 0 && y0 < HH) {
            if (x0 >= 0 && x0 < WW)         acc += w00 * fb[y0 * WW + x0];
            if (x0 + 1 >= 0 && x0 + 1 < WW) acc += w01 * fb[y0 * WW + x0 + 1];
        }
        if (y0 + 1 >= 0 && y0 + 1 < HH) {
            if (x0 >= 0 && x0 < WW)         acc += w10 * fb[(y0 + 1) * WW + x0];
            if (x0 + 1 >= 0 && x0 + 1 < WW) acc += w11 * fb[(y0 + 1) * WW + x0 + 1];
        }
        f2wp[((size_t)(b * CC + c) * (HH + 6) + (y + 3)) * (WW + 6) + (x + 3)] = acc;
    }
}

// ---------------- correlation-volume pack (per chunk of 14 items) ----------------
__global__ void mvol_pack_kernel(const float* __restrict__ f2wp, const float* __restrict__ fmap1,
                                 unsigned short* __restrict__ mv, int n0)
{
    int tid = blockIdx.x * blockDim.x + threadIdx.x;
    if (tid >= NCH * HWSZ) return;
    int x = tid % WW; int y = (tid / WW) % HH; int ni = tid / HWSZ;
    int ng = n0 + ni;
    int b = ng / NDISP; int d = ng % NDISP;
    int di = d / DUU - RR;   // x shift
    int dj = d % DUU - RR;   // y shift
    float s[CC]; float sum = 0.0f;
#pragma unroll
    for (int c = 0; c < CC; ++c) {
        s[c] = f2wp[((size_t)(b * CC + c) * (HH + 6) + (y + dj + 3)) * (WW + 6) + (x + di + 3)];
        sum += s[c];
    }
    float valid = (sum != 0.0f) ? 1.0f : 0.0f;
    size_t base = (((size_t)ni * (HH + 2) + y + 1) * (WW + 2) + x + 1) * 64;
    size_t p = (size_t)y * WW + x;
#pragma unroll
    for (int c = 0; c < CC; ++c) {
        mv[base + c]      = f2bf(fmap1[(size_t)(b * CC + c) * HWSZ + p] * valid);
        mv[base + CC + c] = f2bf(s[c] * valid);
    }
}

// ---------------- displacement-aware projection (49x49 1x1) ----------------
__global__ void dap_kernel(const float* __restrict__ cvol, const float* __restrict__ dw,
                           const float* __restrict__ db, float* __restrict__ cvol2)
{
    int tid = blockIdx.x * blockDim.x + threadIdx.x;
    if (tid >= BB * NDISP * HWSZ) return;
    int p = tid % HWSZ; int o = (tid / HWSZ) % NDISP; int b = tid / (HWSZ * NDISP);
    float a = db[o];
    for (int k = 0; k < NDISP; ++k)
        a += dw[o * NDISP + k] * cvol[(size_t)(b * NDISP + k) * HWSZ + p];
    cvol2[tid] = a;
}

// ---------------- pack kernels ----------------
__global__ void pack_mein_kernel(const float* __restrict__ cvol2, const float* __restrict__ fmap1,
                                 const float* __restrict__ flow, unsigned short* __restrict__ mein)
{
    int tid = blockIdx.x * blockDim.x + threadIdx.x;
    if (tid >= BB * HWSZ) return;
    int x = tid % WW; int y = (tid / WW) % HH; int b = tid / HWSZ;
    size_t p = (size_t)y * WW + x;
    size_t base = (((size_t)b * (HH + 2) + y + 1) * (WW + 2) + x + 1) * 96;
    for (int k = 0; k < NDISP; ++k)
        mein[base + k] = f2bf(cvol2[(size_t)(b * NDISP + k) * HWSZ + p]);
    for (int c = 0; c < CC; ++c)
        mein[base + 49 + c] = f2bf(fmap1[(size_t)(b * CC + c) * HWSZ + p]);
    mein[base + 81] = f2bf(flow[(size_t)(b * 2 + 0) * HWSZ + p]);
    mein[base + 82] = f2bf(flow[(size_t)(b * 2 + 1) * HWSZ + p]);
    // 83..95 stay zero (memset)
}

__global__ void pack_hx1_kernel(const float* __restrict__ hcur, const float* __restrict__ flow,
                                unsigned short* __restrict__ hx1)
{
    int tid = blockIdx.x * blockDim.x + threadIdx.x;
    if (tid >= BB * HWSZ) return;
    int x = tid % WW; int y = (tid / WW) % HH; int b = tid / HWSZ;
    size_t p = (size_t)y * WW + x;
    size_t base = (((size_t)b * HH + y) * (WW + 4) + x + 2) * 224;
    for (int c = 0; c < HIDC; ++c)
        hx1[base + c] = f2bf(hcur[(size_t)(b * HIDC + c) * HWSZ + p]);
    hx1[base + 222] = f2bf(flow[(size_t)(b * 2 + 0) * HWSZ + p]);
    hx1[base + 223] = f2bf(flow[(size_t)(b * 2 + 1) * HWSZ + p]);
}

__global__ void pack_rhx1_kernel(const float* __restrict__ hcur, const float* __restrict__ rg,
                                 const unsigned short* __restrict__ hx1, unsigned short* __restrict__ rhx1)
{
    int tid = blockIdx.x * blockDim.x + threadIdx.x;
    if (tid >= BB * HWSZ) return;
    int x = tid % WW; int y = (tid / WW) % HH; int b = tid / HWSZ;
    size_t p = (size_t)y * WW + x;
    size_t base = (((size_t)b * HH + y) * (WW + 4) + x + 2) * 224;
    for (int c = 0; c < HIDC; ++c)
        rhx1[base + c] = f2bf(rg[(size_t)(b * HIDC + c) * HWSZ + p] *
                              hcur[(size_t)(b * HIDC + c) * HWSZ + p]);
    for (int c = HIDC; c < 224; ++c) rhx1[base + c] = hx1[base + c];
}

__global__ void pack_hx2_kernel(const float* __restrict__ hcur, const unsigned short* __restrict__ hx1,
                                unsigned short* __restrict__ hx2)
{
    int tid = blockIdx.x * blockDim.x + threadIdx.x;
    if (tid >= BB * HWSZ) return;
    int x = tid % WW; int y = (tid / WW) % HH; int b = tid / HWSZ;
    size_t p = (size_t)y * WW + x;
    size_t b2 = (((size_t)b * (HH + 4) + y + 2) * WW + x) * 224;
    size_t b1 = (((size_t)b * HH + y) * (WW + 4) + x + 2) * 224;
    for (int c = 0; c < HIDC; ++c)
        hx2[b2 + c] = f2bf(hcur[(size_t)(b * HIDC + c) * HWSZ + p]);
    for (int c = HIDC; c < 224; ++c) hx2[b2 + c] = hx1[b1 + c];
}

__global__ void pack_rhx2_kernel(const float* __restrict__ hcur, const float* __restrict__ rg,
                                 const unsigned short* __restrict__ hx2, unsigned short* __restrict__ rhx2)
{
    int tid = blockIdx.x * blockDim.x + threadIdx.x;
    if (tid >= BB * HWSZ) return;
    int x = tid % WW; int y = (tid / WW) % HH; int b = tid / HWSZ;
    size_t p = (size_t)y * WW + x;
    size_t b2 = (((size_t)b * (HH + 4) + y + 2) * WW + x) * 224;
    for (int c = 0; c < HIDC; ++c)
        rhx2[b2 + c] = f2bf(rg[(size_t)(b * HIDC + c) * HWSZ + p] *
                            hcur[(size_t)(b * HIDC + c) * HWSZ + p]);
    for (int c = HIDC; c < 224; ++c) rhx2[b2 + c] = hx2[b2 + c];
}

__global__ void gru_update_kernel(float* __restrict__ h, const float* __restrict__ z,
                                  const float* __restrict__ q, int n)
{
    int tid = blockIdx.x * blockDim.x + threadIdx.x;
    if (tid >= n) return;
    float hv = h[tid], zv = z[tid];
    h[tid] = (1.0f - zv) * hv + zv * q[tid];
}

__global__ void pack_hf_kernel(const float* __restrict__ hcur, unsigned short* __restrict__ hf)
{
    int tid = blockIdx.x * blockDim.x + threadIdx.x;
    if (tid >= BB * HWSZ) return;
    int x = tid % WW; int y = (tid / WW) % HH; int b = tid / HWSZ;
    size_t p = (size_t)y * WW + x;
    size_t base = (((size_t)b * HH + y) * WW + x) * (size_t)HIDC;
    for (int c = 0; c < HIDC; ++c)
        hf[base + c] = f2bf(hcur[(size_t)(b * HIDC + c) * HWSZ + p]);
}

__global__ void softmax_flow_kernel(const float* __restrict__ s, const float* __restrict__ flow,
                                    float* __restrict__ out_flow)
{
    int tid = blockIdx.x * blockDim.x + threadIdx.x;
    if (tid >= BB * HWSZ) return;
    int p = tid % HWSZ; int b = tid / HWSZ;
    float vals[NDISP]; float m = -1e30f;
#pragma unroll
    for (int k = 0; k < NDISP; ++k) {
        vals[k] = s[(size_t)(b * NDISP + k) * HWSZ + p];
        m = fmaxf(m, vals[k]);
    }
    float sum = 0.0f;
#pragma unroll
    for (int k = 0; k < NDISP; ++k) { vals[k] = __expf(vals[k] - m); sum += vals[k]; }
    float inv = 1.0f / sum;
    float d0 = 0.0f, d1 = 0.0f;
#pragma unroll
    for (int k = 0; k < NDISP; ++k) {
        float pr = vals[k] * inv;
        d0 += pr * (float)(k / DUU - RR);
        d1 += pr * (float)(k % DUU - RR);
    }
    out_flow[(size_t)(b * 2 + 0) * HWSZ + p] = flow[(size_t)(b * 2 + 0) * HWSZ + p] + d0;
    out_flow[(size_t)(b * 2 + 1) * HWSZ + p] = flow[(size_t)(b * 2 + 1) * HWSZ + p] + d1;
}

// ============================================================
// host side
// ============================================================
static unsigned long long packnib_host(const int* v, int n) {
    unsigned long long p = 0;
    for (int i = 0; i < n; ++i)
        p |= ((unsigned long long)((unsigned)v[i] & 0xFu)) << (4 * i);
    return p;
}

static void conv_dispatch(hipStream_t stream,
                          const void* act, const void* wgt, const float* bias, void* out,
                          int N, int Hp, int Wp, int Cin, int Ho, int Wo, int stride,
                          int iy0, int ix0, int Opad, int Oreal, int T,
                          unsigned long long dys, unsigned long long dxs, int act_code,
                          int out_mode, int oHp, int oWp, int oCt,
                          int oy0, int ox0, int oys, int oxs, int ocoff)
{
    dim3 grid(N * Ho * (Wo >> 4));
    dim3 blockd(256);
    const unsigned short* a = (const unsigned short*)act;
    const unsigned short* w = (const unsigned short*)wgt;
#define CONV_CASE(CIN_)                                                              \
    case CIN_:                                                                       \
        conv_wmma_kernel<CIN_><<<grid, blockd, 0, stream>>>(                         \
            a, w, bias, out, N, Hp, Wp, Ho, Wo, stride, iy0, ix0, Opad, Oreal, T,    \
            dys, dxs, act_code, out_mode, oHp, oWp, oCt, oy0, ox0, oys, oxs, ocoff); \
        break;
    switch (Cin) {
        CONV_CASE(32)
        CONV_CASE(64)
        CONV_CASE(96)
        CONV_CASE(128)
        CONV_CASE(224)
        CONV_CASE(256)
        default: break;
    }
#undef CONV_CASE
}

extern "C" void kernel_launch(void* const* d_in, const int* in_sizes, int n_in,
                              void* d_out, int out_size, void* d_ws, size_t ws_size,
                              hipStream_t stream)
{
    (void)in_sizes; (void)n_in; (void)out_size; (void)ws_size;
    const float* fmap1 = (const float*)d_in[0];
    const float* fmap2 = (const float*)d_in[1];
    const float* h_in  = (const float*)d_in[2];
    const float* flow  = (const float*)d_in[3];
    const float* P[36];
    for (int i = 0; i < 36; ++i) P[i] = (const float*)d_in[4 + i];
    enum { mn_w1, mn_b1, mn_w2, mn_b2, mn_w3, mn_b3, mn_w4, mn_b4, mn_wd, mn_bd,
           mn_w5, mn_b5, dap_w, dap_b, me_w1, me_b1, me_w2, me_b2, me_w3, me_b3,
           gz1_w, gz1_b, gr1_w, gr1_b, gq1_w, gq1_b, gz2_w, gz2_b, gr2_w, gr2_b,
           gq2_w, gq2_b, fh_w1, fh_b1, fh_w2, fh_b2 };

    // ---------------- workspace carve-up ----------------
    char* base = (char*)d_ws; size_t off = 0;
    auto take = [&](size_t bytes) -> void* {
        off = (off + 255) & ~(size_t)255;
        void* p = base + off; off += bytes; return p;
    };
    const size_t f2wp_sz = (size_t)BB * CC * (HH + 6) * (WW + 6) * 4;
    float* f2wp = (float*)take(f2wp_sz);

    unsigned short* Wmn1 = (unsigned short*)take((size_t)9 * 96 * 64 * 2);
    unsigned short* Wmn2 = (unsigned short*)take((size_t)9 * 128 * 96 * 2);
    unsigned short* Wmn3 = (unsigned short*)take((size_t)9 * 128 * 128 * 2);
    unsigned short* Wmn4 = (unsigned short*)take((size_t)9 * 64 * 128 * 2);
    unsigned short* Wdp[4];
    for (int p = 0; p < 4; ++p) Wdp[p] = (unsigned short*)take((size_t)4 * 32 * 64 * 2);
    unsigned short* Wmn5 = (unsigned short*)take((size_t)9 * 16 * 32 * 2);
    unsigned short* Wme1 = (unsigned short*)take((size_t)9 * 128 * 96 * 2);
    unsigned short* Wme2 = (unsigned short*)take((size_t)9 * 128 * 128 * 2);
    unsigned short* Wme3 = (unsigned short*)take((size_t)9 * 96 * 128 * 2);
    unsigned short* Wg[6];
    for (int g = 0; g < 6; ++g) Wg[g] = (unsigned short*)take((size_t)5 * 128 * 224 * 2);
    unsigned short* Wfh1 = (unsigned short*)take((size_t)1 * 256 * 128 * 2);
    unsigned short* Wfh2 = (unsigned short*)take((size_t)1 * 64 * 256 * 2);

    const size_t mvol_sz = (size_t)NCH * (HH + 2) * (WW + 2) * 64 * 2;
    const size_t x1_sz   = (size_t)NCH * (HH + 2) * (WW + 2) * 96 * 2;
    const size_t x2_sz   = (size_t)NCH * 34 * 50 * 128 * 2;
    const size_t x4_sz   = (size_t)NCH * 34 * 50 * 64 * 2;
    const size_t xd_sz   = (size_t)NCH * (HH + 2) * (WW + 2) * 32 * 2;
    unsigned short* mvolA = (unsigned short*)take(mvol_sz);
    unsigned short* x1b   = (unsigned short*)take(x1_sz);
    unsigned short* x2b   = (unsigned short*)take(x2_sz);
    unsigned short* x3b   = (unsigned short*)take(x2_sz);
    unsigned short* x4b   = (unsigned short*)take(x4_sz);
    unsigned short* xdb   = (unsigned short*)take(xd_sz);

    const size_t cvol_sz = (size_t)BB * NDISP * HWSZ * 4;
    float* cvol  = (float*)take(cvol_sz);
    float* cvol2 = (float*)take(cvol_sz);

    const size_t mein_sz = (size_t)BB * (HH + 2) * (WW + 2) * 96 * 2;
    const size_t me1_sz  = (size_t)BB * (HH + 2) * (WW + 2) * 128 * 2;
    unsigned short* meinb = (unsigned short*)take(mein_sz);
    unsigned short* me1b  = (unsigned short*)take(me1_sz);
    unsigned short* me2b  = (unsigned short*)take(me1_sz);

    const size_t hx1_sz = (size_t)BB * HH * (WW + 4) * 224 * 2;
    const size_t hx2_sz = (size_t)BB * (HH + 4) * WW * 224 * 2;
    unsigned short* hx1b  = (unsigned short*)take(hx1_sz);
    unsigned short* rhx1b = (unsigned short*)take(hx1_sz);
    unsigned short* hx2b  = (unsigned short*)take(hx2_sz);
    unsigned short* rhx2b = (unsigned short*)take(hx2_sz);

    const size_t plane_sz = (size_t)BB * HIDC * HWSZ * 4;
    float* zbuf = (float*)take(plane_sz);
    float* rbuf = (float*)take(plane_sz);
    float* qbuf = (float*)take(plane_sz);
    float* hcur = (float*)take(plane_sz);

    unsigned short* hFb = (unsigned short*)take((size_t)BB * HWSZ * 128 * 2);
    unsigned short* s1b = (unsigned short*)take((size_t)BB * HWSZ * 256 * 2);
    float* sbuf = (float*)take((size_t)BB * NDISP * HWSZ * 4);

    // ---------------- helpers ----------------
    auto conv = [&](const void* act, const void* wgt, const float* bias, void* out,
                    int N, int Hp, int Wp, int Cin, int Ho, int Wo, int stride,
                    int iy0, int ix0, int Opad, int Oreal, int T,
                    unsigned long long dys, unsigned long long dxs, int act_code,
                    int out_mode, int oHp, int oWp, int oCt,
                    int oy0, int ox0, int oys, int oxs, int ocoff) {
        conv_dispatch(stream, act, wgt, bias, out, N, Hp, Wp, Cin, Ho, Wo, stride,
                      iy0, ix0, Opad, Oreal, T, dys, dxs, act_code, out_mode,
                      oHp, oWp, oCt, oy0, ox0, oys, oxs, ocoff);
    };
    auto repack = [&](const float* w, unsigned short* dst, int O, int I, int KH, int KW,
                      int Opad, int Ipad, int T, unsigned long long khs, unsigned long long kws) {
        int tot = T * Opad * Ipad;
        repack_w_kernel<<<(tot + 255) / 256, 256, 0, stream>>>(
            w, dst, O, I, KH, KW, Opad, Ipad, T, khs, kws, tot);
    };

    // tap tables
    int dy33[9], dx33[9], kh33[9], kw33[9];
    for (int t = 0; t < 9; ++t) {
        dy33[t] = t / 3 - 1; dx33[t] = t % 3 - 1; kh33[t] = t / 3; kw33[t] = t % 3;
    }
    unsigned long long DYS33 = packnib_host(dy33, 9), DXS33 = packnib_host(dx33, 9);
    unsigned long long KHS33 = packnib_host(kh33, 9), KWS33 = packnib_host(kw33, 9);
    int d15[5] = { -2, -1, 0, 1, 2 }, k15[5] = { 0, 1, 2, 3, 4 }, z5[5] = { 0, 0, 0, 0, 0 };
    unsigned long long DPM2 = packnib_host(d15, 5), KIDX5 = packnib_host(k15, 5), ZERO5 = packnib_host(z5, 5);
    unsigned long long ZERO1 = 0ull;

    // ---------------- weight repack (weights are inputs; redo every launch) ----------------
    repack(P[mn_w1], Wmn1, 96, 64, 3, 3, 96, 64, 9, KHS33, KWS33);
    repack(P[mn_w2], Wmn2, 128, 96, 3, 3, 128, 96, 9, KHS33, KWS33);
    repack(P[mn_w3], Wmn3, 128, 128, 3, 3, 128, 128, 9, KHS33, KWS33);
    repack(P[mn_w4], Wmn4, 64, 128, 3, 3, 64, 128, 9, KHS33, KWS33);
    repack(P[mn_w5], Wmn5, 1, 32, 3, 3, 16, 32, 9, KHS33, KWS33);
    repack(P[me_w1], Wme1, 128, 83, 3, 3, 128, 96, 9, KHS33, KWS33);
    repack(P[me_w2], Wme2, 128, 128, 3, 3, 128, 128, 9, KHS33, KWS33);
    repack(P[me_w3], Wme3, 94, 128, 3, 3, 96, 128, 9, KHS33, KWS33);
    repack(P[gz1_w], Wg[0], 128, 224, 1, 5, 128, 224, 5, ZERO5, KIDX5);
    repack(P[gr1_w], Wg[1], 128, 224, 1, 5, 128, 224, 5, ZERO5, KIDX5);
    repack(P[gq1_w], Wg[2], 128, 224, 1, 5, 128, 224, 5, ZERO5, KIDX5);
    repack(P[gz2_w], Wg[3], 128, 224, 5, 1, 128, 224, 5, KIDX5, ZERO5);
    repack(P[gr2_w], Wg[4], 128, 224, 5, 1, 128, 224, 5, KIDX5, ZERO5);
    repack(P[gq2_w], Wg[5], 128, 224, 5, 1, 128, 224, 5, KIDX5, ZERO5);
    repack(P[fh_w1], Wfh1, 256, 128, 1, 1, 256, 128, 1, ZERO1, ZERO1);
    repack(P[fh_w2], Wfh2, 49, 256, 1, 1, 64, 256, 1, ZERO1, ZERO1);
    // deconv (k=4,s=2,p=1) -> 4 parity convs of 2x2 taps
    unsigned long long DYSp[4], DXSp[4];
    for (int par = 0; par < 4; ++par) {
        int py = par >> 1, px = par & 1;
        int khl[2] = { py ? 1 : 0, py ? 3 : 2 }, dyl[2] = { py ? 0 : -1, py ? 1 : 0 };
        int kwl[2] = { px ? 1 : 0, px ? 3 : 2 }, dxl[2] = { px ? 0 : -1, px ? 1 : 0 };
        int kh4[4], kw4[4], dy4[4], dx4[4];
        for (int t = 0; t < 4; ++t) {
            kh4[t] = khl[t / 2]; kw4[t] = kwl[t % 2];
            dy4[t] = dyl[t / 2]; dx4[t] = dxl[t % 2];
        }
        DYSp[par] = packnib_host(dy4, 4); DXSp[par] = packnib_host(dx4, 4);
        repack(P[mn_wd], Wdp[par], 32, 64, 4, 4, 32, 64, 4,
               packnib_host(kh4, 4), packnib_host(kw4, 4));
    }

    // ---------------- warp + init ----------------
    hipMemsetAsync(f2wp, 0, f2wp_sz, stream);
    warp_pack_kernel<<<(BB * HWSZ + 255) / 256, 256, 0, stream>>>(fmap2, flow, f2wp);
    hipMemcpyAsync(hcur, h_in, plane_sz, hipMemcpyDeviceToDevice, stream);

    // ---------------- matching net, 7 chunks of 14 items ----------------
    for (int ch = 0; ch < NCHUNKS; ++ch) {
        int n0 = ch * NCH;
        hipMemsetAsync(mvolA, 0, mvol_sz, stream);
        hipMemsetAsync(x1b, 0, x1_sz, stream);
        hipMemsetAsync(x2b, 0, x2_sz, stream);
        hipMemsetAsync(x3b, 0, x2_sz, stream);
        hipMemsetAsync(x4b, 0, x4_sz, stream);
        hipMemsetAsync(xdb, 0, xd_sz, stream);
        mvol_pack_kernel<<<(NCH * HWSZ + 255) / 256, 256, 0, stream>>>(f2wp, fmap1, mvolA, n0);
        // conv1: 64 -> 96, relu
        conv(mvolA, Wmn1, P[mn_b1], x1b, NCH, 66, 98, 64, 64, 96, 1, 1, 1,
             96, 96, 9, DYS33, DXS33, 1, 0, 66, 98, 96, 1, 1, 1, 1, 0);
        // conv2: 96 -> 128, stride 2, relu
        conv(x1b, Wmn2, P[mn_b2], x2b, NCH, 66, 98, 96, 32, 48, 2, 1, 1,
             128, 128, 9, DYS33, DXS33, 1, 0, 34, 50, 128, 1, 1, 1, 1, 0);
        // conv3: 128 -> 128, relu
        conv(x2b, Wmn3, P[mn_b3], x3b, NCH, 34, 50, 128, 32, 48, 1, 1, 1,
             128, 128, 9, DYS33, DXS33, 1, 0, 34, 50, 128, 1, 1, 1, 1, 0);
        // conv4: 128 -> 64, relu
        conv(x3b, Wmn4, P[mn_b4], x4b, NCH, 34, 50, 128, 32, 48, 1, 1, 1,
             64, 64, 9, DYS33, DXS33, 1, 0, 34, 50, 64, 1, 1, 1, 1, 0);
        // deconv as 4 parity convs: 64 -> 32, relu, stride-2 scatter store
        for (int par = 0; par < 4; ++par) {
            int py = par >> 1, px = par & 1;
            conv(x4b, Wdp[par], P[mn_bd], xdb, NCH, 34, 50, 64, 32, 48, 1, 1, 1,
                 32, 32, 4, DYSp[par], DXSp[par], 1, 0, 66, 98, 32, 1 + py, 1 + px, 2, 2, 0);
        }
        // conv5: 32 -> 1, no act, f32 out into cvol[n0..n0+13]
        conv(xdb, Wmn5, P[mn_b5], cvol + (size_t)n0 * HWSZ, NCH, 66, 98, 32, 64, 96, 1, 1, 1,
             16, 1, 9, DYS33, DXS33, 0, 1, 64, 96, 1, 0, 0, 1, 1, 0);
    }

    // ---------------- DAP (49x49 1x1) ----------------
    dap_kernel<<<(BB * NDISP * HWSZ + 255) / 256, 256, 0, stream>>>(cvol, P[dap_w], P[dap_b], cvol2);

    // ---------------- motion encoder ----------------
    hipMemsetAsync(meinb, 0, mein_sz, stream);
    hipMemsetAsync(me1b, 0, me1_sz, stream);
    hipMemsetAsync(me2b, 0, me1_sz, stream);
    hipMemsetAsync(hx1b, 0, hx1_sz, stream);
    pack_mein_kernel<<<(BB * HWSZ + 255) / 256, 256, 0, stream>>>(cvol2, fmap1, flow, meinb);
    conv(meinb, Wme1, P[me_b1], me1b, BB, 66, 98, 96, 64, 96, 1, 1, 1,
         128, 128, 9, DYS33, DXS33, 2, 0, 66, 98, 128, 1, 1, 1, 1, 0);
    conv(me1b, Wme2, P[me_b2], me2b, BB, 66, 98, 128, 64, 96, 1, 1, 1,
         128, 128, 9, DYS33, DXS33, 2, 0, 66, 98, 128, 1, 1, 1, 1, 0);
    pack_hx1_kernel<<<(BB * HWSZ + 255) / 256, 256, 0, stream>>>(hcur, flow, hx1b);
    // me3: 128 -> 94 (no act), store into hx1 channels 128..221
    conv(me2b, Wme3, P[me_b3], hx1b, BB, 66, 98, 128, 64, 96, 1, 1, 1,
         96, 94, 9, DYS33, DXS33, 0, 0, 64, 100, 224, 0, 2, 1, 1, 128);

    // ---------------- GRU pass 1 (1x5 horizontal) ----------------
    conv(hx1b, Wg[0], P[gz1_b], zbuf, BB, 64, 100, 224, 64, 96, 1, 0, 2,
         128, 128, 5, ZERO5, DPM2, 3, 1, 64, 96, 128, 0, 0, 1, 1, 0);
    conv(hx1b, Wg[1], P[gr1_b], rbuf, BB, 64, 100, 224, 64, 96, 1, 0, 2,
         128, 128, 5, ZERO5, DPM2, 3, 1, 64, 96, 128, 0, 0, 1, 1, 0);
    hipMemsetAsync(rhx1b, 0, hx1_sz, stream);
    pack_rhx1_kernel<<<(BB * HWSZ + 255) / 256, 256, 0, stream>>>(hcur, rbuf, hx1b, rhx1b);
    conv(rhx1b, Wg[2], P[gq1_b], qbuf, BB, 64, 100, 224, 64, 96, 1, 0, 2,
         128, 128, 5, ZERO5, DPM2, 4, 1, 64, 96, 128, 0, 0, 1, 1, 0);
    gru_update_kernel<<<(BB * HIDC * HWSZ + 255) / 256, 256, 0, stream>>>(
        hcur, zbuf, qbuf, BB * HIDC * HWSZ);

    // ---------------- GRU pass 2 (5x1 vertical) ----------------
    hipMemsetAsync(hx2b, 0, hx2_sz, stream);
    pack_hx2_kernel<<<(BB * HWSZ + 255) / 256, 256, 0, stream>>>(hcur, hx1b, hx2b);
    conv(hx2b, Wg[3], P[gz2_b], zbuf, BB, 68, 96, 224, 64, 96, 1, 2, 0,
         128, 128, 5, DPM2, ZERO5, 3, 1, 64, 96, 128, 0, 0, 1, 1, 0);
    conv(hx2b, Wg[4], P[gr2_b], rbuf, BB, 68, 96, 224, 64, 96, 1, 2, 0,
         128, 128, 5, DPM2, ZERO5, 3, 1, 64, 96, 128, 0, 0, 1, 1, 0);
    hipMemsetAsync(rhx2b, 0, hx2_sz, stream);
    pack_rhx2_kernel<<<(BB * HWSZ + 255) / 256, 256, 0, stream>>>(hcur, rbuf, hx2b, rhx2b);
    conv(rhx2b, Wg[5], P[gq2_b], qbuf, BB, 68, 96, 224, 64, 96, 1, 2, 0,
         128, 128, 5, DPM2, ZERO5, 4, 1, 64, 96, 128, 0, 0, 1, 1, 0);
    gru_update_kernel<<<(BB * HIDC * HWSZ + 255) / 256, 256, 0, stream>>>(
        hcur, zbuf, qbuf, BB * HIDC * HWSZ);

    // ---------------- flow head ----------------
    pack_hf_kernel<<<(BB * HWSZ + 255) / 256, 256, 0, stream>>>(hcur, hFb);
    conv(hFb, Wfh1, P[fh_b1], s1b, BB, 64, 96, 128, 64, 96, 1, 0, 0,
         256, 256, 1, ZERO1, ZERO1, 2, 0, 64, 96, 256, 0, 0, 1, 1, 0);
    conv(s1b, Wfh2, P[fh_b2], sbuf, BB, 64, 96, 256, 64, 96, 1, 0, 0,
         64, 49, 1, ZERO1, ZERO1, 2, 1, 64, 96, 49, 0, 0, 1, 1, 0);

    // ---------------- outputs: (h, flow + d) ----------------
    float* out_h = (float*)d_out;
    float* out_flow = out_h + (size_t)BB * HIDC * HWSZ;
    softmax_flow_kernel<<<(BB * HWSZ + 255) / 256, 256, 0, stream>>>(sbuf, flow, out_flow);
    hipMemcpyAsync(out_h, hcur, plane_sz, hipMemcpyDeviceToDevice, stream);
}
